// SupervisedGraphsage_24137716203619
// MI455X (gfx1250) — matrix operand order repaired and checked by reading, not compiled, not run on hardware
//
#include <hip/hip_runtime.h>
#include <hip/hip_bf16.h>
#include <cstddef>

// ---------------------------------------------------------------------------
// GraphSAGE forward for MI455X (gfx1250, wave32, WMMA).
//
// Roofline: f2 gather = 256000*512*4B = 524MB dominates (≈23us @ 23.3TB/s);
// GEMM flops ≈ 7.4G are trivial at bf16-WMMA rates -> fuse gather+mean into
// LDS tiles, run GEMMs with v_wmma_f32_16x16x32_bf16, keep f32 accumulators.
// ---------------------------------------------------------------------------

typedef __attribute__((ext_vector_type(16))) __bf16 v16bf;
typedef __attribute__((ext_vector_type(8)))  __bf16 v8bf;
typedef __attribute__((ext_vector_type(8)))  float  v8f;

#define WMMA_BF16(a, b, c) \
  __builtin_amdgcn_wmma_f32_16x16x32_bf16(false, (a), false, (b), (short)0, (c), false, false)

static __device__ __forceinline__ v16bf cat8(v8bf lo, v8bf hi) {
  return __builtin_shufflevector(lo, hi, 0,1,2,3,4,5,6,7,8,9,10,11,12,13,14,15);
}

// A-fragment (16x32 bf16, ISA 7.12.2): lane m(0..15) holds row m.
// elems 0..15 -> K = k0+16*hi .. +15 (matches B packing: e + 16*hi)
// Both runs are 16B-aligned in LDS (row pitch is a multiple of 8 halfs).
static __device__ __forceinline__ v16bf load_a_frag(const __bf16* row, int k0, int hi) {
  v8bf lo = *(const v8bf*)(row + k0 + 16 * hi);
  v8bf hh = *(const v8bf*)(row + k0 + 16 * hi + 8);
  return cat8(lo, hh);
}

// ---------------------------------------------------------------------------
// Weight pre-swizzle: W[K,N] f32 (row-major) -> bf16 fragments laid out so a
// lane's 16 B elements are contiguous (two global_load_b128 in the GEMM loop).
// B-fragment layout (32x16, K x N): lane n = lane&15; lanes 0-15 hold K 0..15,
// lanes 16-31 hold K 16..31 (elem e -> K = e + 16*hi). N padded with zeros.
// out[((ks*NT + nt)*32 + lane)*16 + e]
// ---------------------------------------------------------------------------
__global__ __launch_bounds__(256) void k_prep_w(const float* __restrict__ W,
                                                __bf16* __restrict__ out,
                                                int N, int NT) {
  int idx  = blockIdx.x * 256 + threadIdx.x;
  int e    = idx & 15;
  int lane = (idx >> 4) & 31;
  int til  = idx >> 9;
  int nt   = til % NT;
  int ks   = til / NT;
  int hi   = lane >> 4;
  int kk   = (ks << 5) + e + (hi << 4);
  int n    = (nt << 4) + (lane & 15);
  float v  = (n < N) ? W[(size_t)kk * N + n] : 0.0f;
  out[idx] = (__bf16)v;
}

// ---------------------------------------------------------------------------
// Layer 2: h2[25600,256] = relu([f1@W2s+b2s | agg2@W2n+b2n])
// agg2[r,d] = mean over 10 consecutive elems of the flattened 10-row gather
// block (faithful to torch's view(...,D,n2).mean(-1) quirk).
// 1600 blocks x 256 thr (8 waves). Block = 16 M-rows; wave w -> N-tile w of
// both GEMMs. K=512 -> 16 k-steps of 32.
// ---------------------------------------------------------------------------
__global__ __launch_bounds__(256) void k_h2(const float* __restrict__ feat,
                                            const int* __restrict__ ids1,
                                            const int* __restrict__ ids2,
                                            const __bf16* __restrict__ W2s_p,
                                            const float* __restrict__ b2s,
                                            const __bf16* __restrict__ W2n_p,
                                            const float* __restrict__ b2n,
                                            float* __restrict__ h2) {
  __shared__ __bf16 sA1[16][520];           // f1 rows   (pitch 520*2B = 16B-aligned)
  __shared__ __bf16 sA2[16][520];           // agg2 rows
  __shared__ unsigned long long sOff[16][10];

  const int tid  = threadIdx.x;
  const int row0 = blockIdx.x * 16;

  // Stage gather-block base offsets (160 feature rows feeding agg2).
  if (tid < 160) {
    int r = tid / 10, j = tid % 10;
    sOff[r][j] = (unsigned long long)ids2[(size_t)(row0 + r) * 10 + j] * 512ull;
  }

  // Gather f1 rows -> bf16 LDS (coalesced float4 stream).
#pragma unroll
  for (int i = 0; i < 8; ++i) {
    int idx = tid + 256 * i;
    int r = idx >> 7, c = (idx & 127) << 2;
    float4 v = *(const float4*)(feat + (size_t)ids1[row0 + r] * 512 + c);
    sA1[r][c + 0] = (__bf16)v.x; sA1[r][c + 1] = (__bf16)v.y;
    sA1[r][c + 2] = (__bf16)v.z; sA1[r][c + 3] = (__bf16)v.w;
  }
  __syncthreads();

  // Fused gather + quirky mean: agg2[r][d] = 0.1 * sum_{k<10} flat[10d+k],
  // flat[t] = feat[ids2[10R + t/512]][t%512]. Adjacent threads read adjacent
  // 10-float chunks -> contiguous 1280B per wave inside the gathered rows.
#pragma unroll 4
  for (int i = 0; i < 32; ++i) {
    int idx = tid + 256 * i;
    int r = idx >> 9, d = idx & 511;
    const unsigned long long* off = sOff[r];
    float s = 0.0f;
    int t = 10 * d;
#pragma unroll
    for (int k = 0; k < 10; ++k) {
      int tt = t + k;
      s += feat[off[tt >> 9] + (tt & 511)];
    }
    sA2[r][d] = (__bf16)(s * 0.1f);
  }
  __syncthreads();

  const int w = tid >> 5, lane = tid & 31, m = lane & 15, hi = lane >> 4;
  v8f accs = {};
  v8f accn = {};
#pragma unroll 2
  for (int ks = 0; ks < 16; ++ks) {
    const int k0 = ks << 5;
    v16bf a1 = load_a_frag(sA1[m], k0, hi);
    v16bf a2 = load_a_frag(sA2[m], k0, hi);
    v16bf bs = *(const v16bf*)(W2s_p + (((size_t)ks * 8 + w) * 32 + lane) * 16);
    v16bf bn = *(const v16bf*)(W2n_p + (((size_t)ks * 8 + w) * 32 + lane) * 16);
    accs = WMMA_BF16(a1, bs, accs);
    accn = WMMA_BF16(a2, bn, accn);
  }

  const int col = 16 * w + m;
  const float bs_ = b2s[col], bn_ = b2n[col];
#pragma unroll
  for (int v = 0; v < 8; ++v) {
    int r = v + 8 * hi;
    float xs = accs[v] + bs_; xs = xs > 0.0f ? xs : 0.0f;
    float xn = accn[v] + bn_; xn = xn > 0.0f ? xn : 0.0f;
    size_t base = (size_t)(row0 + r) * 256;
    h2[base + col]       = xs;
    h2[base + 128 + col] = xn;
  }
}

// ---------------------------------------------------------------------------
// agg1[b,c]: h2's 25 rows per sample are contiguous, so the torch view-quirk
// mean collapses to a flat 25-element contiguous mean.
// ---------------------------------------------------------------------------
__global__ __launch_bounds__(256) void k_agg1(const float* __restrict__ h2,
                                              float* __restrict__ agg1) {
  const int b = blockIdx.x, c = threadIdx.x;
  const float* base = h2 + (size_t)b * 25 * 256;
  float s = 0.0f;
  const int t0 = c * 25;
#pragma unroll
  for (int k = 0; k < 25; ++k) s += base[t0 + k];
  agg1[(size_t)b * 256 + c] = s * 0.04f;
}

// ---------------------------------------------------------------------------
// Layer 1: h1[1024,256] = [f0@W1s+b1s | agg1@W1n+b1n]   (no relu)
// 64 blocks x 256 thr; block = 16 rows; wave w -> N-tile w of both GEMMs.
// ---------------------------------------------------------------------------
__global__ __launch_bounds__(256) void k_h1(const float* __restrict__ feat,
                                            const int* __restrict__ ids0,
                                            const float* __restrict__ agg1,
                                            const __bf16* __restrict__ W1s_p,
                                            const float* __restrict__ b1s,
                                            const __bf16* __restrict__ W1n_p,
                                            const float* __restrict__ b1n,
                                            float* __restrict__ h1) {
  __shared__ __bf16 sA1[16][520];   // f0 rows, K=512
  __shared__ __bf16 sA2[16][264];   // agg1 rows, K=256 (pitch 264*2B 16B-aligned)

  const int tid = threadIdx.x;
  const int row0 = blockIdx.x * 16;

#pragma unroll
  for (int i = 0; i < 8; ++i) {
    int idx = tid + 256 * i;
    int r = idx >> 7, c = (idx & 127) << 2;
    float4 v = *(const float4*)(feat + (size_t)ids0[row0 + r] * 512 + c);
    sA1[r][c + 0] = (__bf16)v.x; sA1[r][c + 1] = (__bf16)v.y;
    sA1[r][c + 2] = (__bf16)v.z; sA1[r][c + 3] = (__bf16)v.w;
  }
#pragma unroll
  for (int i = 0; i < 4; ++i) {
    int idx = tid + 256 * i;
    int r = idx >> 6, c = (idx & 63) << 2;
    float4 v = *(const float4*)(agg1 + (size_t)(row0 + r) * 256 + c);
    sA2[r][c + 0] = (__bf16)v.x; sA2[r][c + 1] = (__bf16)v.y;
    sA2[r][c + 2] = (__bf16)v.z; sA2[r][c + 3] = (__bf16)v.w;
  }
  __syncthreads();

  const int w = tid >> 5, lane = tid & 31, m = lane & 15, hi = lane >> 4;
  v8f accs = {};
  v8f accn = {};
#pragma unroll 2
  for (int ks = 0; ks < 16; ++ks) {
    v16bf a = load_a_frag(sA1[m], ks << 5, hi);
    v16bf b = *(const v16bf*)(W1s_p + (((size_t)ks * 8 + w) * 32 + lane) * 16);
    accs = WMMA_BF16(a, b, accs);
  }
#pragma unroll 2
  for (int ks = 0; ks < 8; ++ks) {
    v16bf a = load_a_frag(sA2[m], ks << 5, hi);
    v16bf b = *(const v16bf*)(W1n_p + (((size_t)ks * 8 + w) * 32 + lane) * 16);
    accn = WMMA_BF16(a, b, accn);
  }

  const int col = 16 * w + m;
  const float bs_ = b1s[col], bn_ = b1n[col];
#pragma unroll
  for (int v = 0; v < 8; ++v) {
    int r = v + 8 * hi;
    size_t base = (size_t)(row0 + r) * 256;
    h1[base + col]       = accs[v] + bs_;
    h1[base + 128 + col] = accn[v] + bn_;
  }
}

// ---------------------------------------------------------------------------
// FC: out[1024,41] = h1 @ Wfc + bfc.  N padded to 48 (3 tiles); waves 0..2
// compute, masked store for cols >= 41.  64 blocks x 128 thr.
// ---------------------------------------------------------------------------
__global__ __launch_bounds__(128) void k_fc(const float* __restrict__ h1,
                                            const __bf16* __restrict__ Wfc_p,
                                            const float* __restrict__ bfc,
                                            float* __restrict__ out) {
  __shared__ __bf16 sA[16][264];
  const int tid = threadIdx.x;
  const int row0 = blockIdx.x * 16;

#pragma unroll
  for (int i = 0; i < 8; ++i) {
    int idx = tid + 128 * i;
    int r = idx >> 6, c = (idx & 63) << 2;
    float4 v = *(const float4*)(h1 + (size_t)(row0 + r) * 256 + c);
    sA[r][c + 0] = (__bf16)v.x; sA[r][c + 1] = (__bf16)v.y;
    sA[r][c + 2] = (__bf16)v.z; sA[r][c + 3] = (__bf16)v.w;
  }
  __syncthreads();

  const int w = tid >> 5, lane = tid & 31, m = lane & 15, hi = lane >> 4;
  if (w < 3) {
    v8f acc = {};
#pragma unroll
    for (int ks = 0; ks < 8; ++ks) {
      v16bf a = load_a_frag(sA[m], ks << 5, hi);
      v16bf b = *(const v16bf*)(Wfc_p + (((size_t)ks * 3 + w) * 32 + lane) * 16);
      acc = WMMA_BF16(a, b, acc);
    }
    const int col = 16 * w + m;
    if (col < 41) {
      const float bias = bfc[col];
#pragma unroll
      for (int v = 0; v < 8; ++v)
        out[(size_t)(row0 + v + 8 * hi) * 41 + col] = acc[v] + bias;
    }
  }
}

// ---------------------------------------------------------------------------
// Host launcher.  Workspace layout (bytes):
//   h2    @ 0           : 25600*256*4 = 26,214,400
//   agg1  @ 26,214,400  :  1024*256*4 =  1,048,576
//   h1    @ 27,262,976  :  1024*256*4 =  1,048,576
//   W2s_p @ 28,311,552  : 65536 bf16, then W2n_p, W1s_p (65536 each),
//   W1n_p (32768), Wfc_p (12288)   -> total ≈ 28.8 MB
// ---------------------------------------------------------------------------
extern "C" void kernel_launch(void* const* d_in, const int* in_sizes, int n_in,
                              void* d_out, int out_size, void* d_ws, size_t ws_size,
                              hipStream_t stream) {
  const float* feat = (const float*)d_in[0];
  const int* ids0 = (const int*)d_in[1];
  const int* ids1 = (const int*)d_in[2];
  const int* ids2 = (const int*)d_in[3];
  const float* W2s = (const float*)d_in[4];  const float* b2s = (const float*)d_in[5];
  const float* W2n = (const float*)d_in[6];  const float* b2n = (const float*)d_in[7];
  const float* W1s = (const float*)d_in[8];  const float* b1s = (const float*)d_in[9];
  const float* W1n = (const float*)d_in[10]; const float* b1n = (const float*)d_in[11];
  const float* Wfc = (const float*)d_in[12]; const float* bfc = (const float*)d_in[13];
  float* out = (float*)d_out;

  char* ws = (char*)d_ws;
  float*  h2    = (float*)(ws);
  float*  agg1  = (float*)(ws + 26214400u);
  float*  h1    = (float*)(ws + 27262976u);
  __bf16* W2s_p = (__bf16*)(ws + 28311552u);
  __bf16* W2n_p = W2s_p + 65536;
  __bf16* W1s_p = W2n_p + 65536;
  __bf16* W1n_p = W1s_p + 65536;
  __bf16* Wfc_p = W1n_p + 32768;

  // Weight fragment pre-swizzle (tiny; re-run each call for determinism).
  k_prep_w<<<256, 256, 0, stream>>>(W2s, W2s_p, 128, 8);   // K=512
  k_prep_w<<<256, 256, 0, stream>>>(W2n, W2n_p, 128, 8);   // K=512
  k_prep_w<<<256, 256, 0, stream>>>(W1s, W1s_p, 128, 8);   // K=512
  k_prep_w<<<128, 256, 0, stream>>>(W1n, W1n_p, 128, 8);   // K=256
  k_prep_w<<< 48, 256, 0, stream>>>(Wfc, Wfc_p,  41, 3);   // K=256, N pad 48

  k_h2  <<<1600, 256, 0, stream>>>(feat, ids1, ids2, W2s_p, b2s, W2n_p, b2n, h2);
  k_agg1<<<1024, 256, 0, stream>>>(h2, agg1);
  k_h1  <<<  64, 256, 0, stream>>>(feat, ids0, agg1, W1s_p, b1s, W1n_p, b1n, h1);
  k_fc  <<<  64, 128, 0, stream>>>(h1, Wfc_p, bfc, out);
}
